// FFT_Linear_46540265619943
// MI455X (gfx1250) — matrix operand order repaired and checked by reading, not compiled, not run on hardware
//
#include <hip/hip_runtime.h>
#include <stdint.h>
#include <stddef.h>

// ---------------- problem constants (from reference) ----------------
#define BLOCK    128
#define OBLK     512                 // output blocks (M rows of the GEMM)
#define KTOT     65536               // K = IN_CH, flattened (j,k)
#define MT       16                  // WMMA M tile
#define MTILES   (OBLK / MT)         // 32
#define KSPLIT   32                  // split-K factor -> 1024 workgroups
#define KRANGE   (KTOT / KSPLIT)     // 2048 K per workgroup -> 64 wmma K-steps
#define CHUNK_BYTES (BLOCK * 32 * 2) // one swizzled B chunk (K=32 x N=128 bf16)

// ---------------- CDNA5 WMMA types ----------------
typedef __attribute__((ext_vector_type(16))) __bf16   v16bf;
typedef __attribute__((ext_vector_type(8)))  float    v8f;
typedef __attribute__((ext_vector_type(4)))  float    vf4;
typedef __attribute__((ext_vector_type(4)))  uint32_t vu4;

union ABFrag { v16bf v; uint32_t u[8]; vu4 q[2]; };

// pack two fp32 -> packed bf16 pair (lo -> [15:0], hi -> [31:16]), RNE.
// gfx1250 has v_cvt_pk_bf16_f32 (confirmed in round-2 codegen); emit the
// packed form directly: 1 VALU per pair instead of clang's cvt+mov_b16 combo.
__device__ __forceinline__ uint32_t pack2_bf16(float lo, float hi) {
  uint32_t r;
  asm("v_cvt_pk_bf16_f32 %0, %1, %2" : "=v"(r) : "v"(lo), "v"(hi));
  return r;
}

// ---------------- kernel 0: zero the output (we accumulate atomically) ---
__global__ __launch_bounds__(256) void zero_out(float* __restrict__ out) {
  out[blockIdx.x * 256 + threadIdx.x] = 0.0f;
}

// ---------------- kernel 1: build circulant X in WMMA-B fragment order ---
// X[m, n] = x[(m & ~127) + ((n - m) & 127)],  m = K index, n = 0..127
// Layout: chunk kkg (K base = kkg*32): [n: 0..127][klocal: 0..31] bf16.
__global__ __launch_bounds__(256) void build_xswz(const float* __restrict__ x,
                                                  uint16_t* __restrict__ xswz) {
  int t   = blockIdx.x * 256 + threadIdx.x;  // t = kkg*128 + n, 2048*128 total
  int kkg = t >> 7;
  int n   = t & 127;
  int m0  = kkg * 32;                        // aligned 32-run never crosses a block
  const float* win = x + (m0 & ~127);
  int c = (n - m0) & 127;
  uint32_t buf[16];
#pragma unroll
  for (int p = 0; p < 16; ++p) {
    float lo = win[(c - 2 * p)     & 127];   // klocal = 2p
    float hi = win[(c - 2 * p - 1) & 127];   // klocal = 2p+1
    buf[p] = pack2_bf16(lo, hi);
  }
  vu4* dst = (vu4*)(xswz + (size_t)t * 32);
  const vu4* s = (const vu4*)buf;
  dst[0] = s[0]; dst[1] = s[1]; dst[2] = s[2]; dst[3] = s[3];
}

// ---------------- kernel 2: split-K GEMM, bf16 WMMA, fp32 accumulate -----
// grid = (MTILES, KSPLIT), 128 threads = 4 waves.
// Wave wv owns N columns [wv*32, wv*32+32) as two 16-col WMMA tiles, so each
// converted A fragment feeds TWO wmmas (conversion cost amortized 2x, and the
// two independent accumulators keep the XDL pipe full).
__global__ __launch_bounds__(128) void circulant_gemm(
    const float* __restrict__ W, const uint16_t* __restrict__ xswz,
    float* __restrict__ out) {
  const int lane  = threadIdx.x & 31;
  const int wv    = threadIdx.x >> 5;  // 0..3
  const int row   = lane & 15;         // A: M row / B: column within tile
  const int h     = lane >> 4;         // lane-half selects K sub-runs
  const int obase = blockIdx.x * MT;
  const int kbase = blockIdx.y * KRANGE;

  // A (fp32 weights, row-major o x K): lane reads runs [h*8, h*8+8) and +16
  const float* wp = W + (size_t)(obase + row) * KTOT + kbase + h * 8;
  // B (pre-swizzled bf16 X): lane reads two 16-byte runs (k h*8..+7, h*8+16..+23) per chunk, per tile
  const char* bp0 = (const char*)xswz + (size_t)(kbase >> 5) * CHUNK_BYTES
                  + (size_t)(wv * 32 + row) * 64 + (size_t)h * 16;
  const char* bp1 = bp0 + 16 * 64;     // next 16-column N tile

  v8f acc0 = {0.f,0.f,0.f,0.f,0.f,0.f,0.f,0.f};
  v8f acc1 = {0.f,0.f,0.f,0.f,0.f,0.f,0.f,0.f};

#pragma unroll 2
  for (int kk = 0; kk < KRANGE / 32; ++kk) {
    ABFrag a, b0, b1;
    // B: regular loads (X lives in L2; keep it resident)
    const vu4* bq0 = (const vu4*)(bp0 + (size_t)kk * CHUNK_BYTES);
    const vu4* bq1 = (const vu4*)(bp1 + (size_t)kk * CHUNK_BYTES);
    b0.q[0] = bq0[0]; b0.q[1] = bq0[2];
    b1.q[0] = bq1[0]; b1.q[1] = bq1[2];

    // A: stream fp32 weights nontemporally (128 MB read exactly once)
    const vf4* aq = (const vf4*)(wp + kk * 32);
    vf4 r0 = __builtin_nontemporal_load(aq + 0);  // K h*8    .. +3
    vf4 r1 = __builtin_nontemporal_load(aq + 1);  // K h*8+4  .. +7
    vf4 r2 = __builtin_nontemporal_load(aq + 4);  // K h*8+16 .. +19
    vf4 r3 = __builtin_nontemporal_load(aq + 5);  // K h*8+20 .. +23

    a.u[0] = pack2_bf16(r0.x, r0.y); a.u[1] = pack2_bf16(r0.z, r0.w);
    a.u[2] = pack2_bf16(r1.x, r1.y); a.u[3] = pack2_bf16(r1.z, r1.w);
    a.u[4] = pack2_bf16(r2.x, r2.y); a.u[5] = pack2_bf16(r2.z, r2.w);
    a.u[6] = pack2_bf16(r3.x, r3.y); a.u[7] = pack2_bf16(r3.z, r3.w);

    acc0 = __builtin_amdgcn_wmma_f32_16x16x32_bf16(
        false, a.v, false, b0.v, (short)0, acc0, false, false);
    acc1 = __builtin_amdgcn_wmma_f32_16x16x32_bf16(
        false, a.v, false, b1.v, (short)0, acc1, false, false);
  }

  // C/D layout: VGPR r -> M = r + h*8, N = lane&15 (within tile)
  float* op0 = out + (size_t)(obase + h * 8) * BLOCK + wv * 32 + row;
  float* op1 = op0 + 16;
#pragma unroll
  for (int r = 0; r < 8; ++r) {
    atomicAdd(op0 + (size_t)r * BLOCK, acc0[r]);
    atomicAdd(op1 + (size_t)r * BLOCK, acc1[r]);
  }
}

// ---------------- launch ----------------
extern "C" void kernel_launch(void* const* d_in, const int* in_sizes, int n_in,
                              void* d_out, int out_size, void* d_ws, size_t ws_size,
                              hipStream_t stream) {
  const float* x = (const float*)d_in[0];          // [65536]
  const float* W = (const float*)d_in[1];          // [512,512,128] row-major
  float* out     = (float*)d_out;                  // [65536]
  uint16_t* xswz = (uint16_t*)d_ws;                // needs 16 MB scratch

  zero_out<<<OBLK * BLOCK / 256, 256, 0, stream>>>(out);
  build_xswz<<<(KTOT / 32) * BLOCK / 256, 256, 0, stream>>>(x, xswz);
  circulant_gemm<<<dim3(MTILES, KSPLIT), 128, 0, stream>>>(W, xswz, out);
}
